// UnifiedAttention_80573586473546
// MI455X (gfx1250) — matrix operand order repaired and checked by reading, not compiled, not run on hardware
//
#include <hip/hip_runtime.h>
#include <math.h>

// ---------------------------------------------------------------------------
// Problem constants
// ---------------------------------------------------------------------------
#define BSZ   64
#define NTOK  197
#define DIM   768
#define NH    12
#define HD    64
#define NP    196            // 14*14
#define HGRID 14
#define ROWS  (BSZ * NTOK)   // 12608
#define CROWS (BSZ * NP)     // 12544
#define OUTN  (ROWS * DIM)   // 9682944
#define SCALE 0.125f
#define ALPHA 0.01f
#define EPSF  1.1920929e-07f
#define BN_EPS 1e-5f

// ---------------------------------------------------------------------------
// WMMA types (CDNA5 / gfx1250, wave32)
// ---------------------------------------------------------------------------
typedef __attribute__((ext_vector_type(16))) __bf16    v16bf;
typedef __attribute__((ext_vector_type(8)))  float     v8f;
typedef __attribute__((ext_vector_type(4)))  unsigned  v4u;

union Frag {
    v16bf v;
    v4u   q[2];
    unsigned short u[16];
};

__device__ __forceinline__ v8f wmma_bf16(Frag a, Frag b, v8f c) {
    // emits v_wmma_f32_16x16x32_bf16
    return __builtin_amdgcn_wmma_f32_16x16x32_bf16(
        /*neg_a=*/false, a.v, /*neg_b=*/false, b.v,
        /*c_mod=*/(short)0, c, /*reuse_a=*/false, /*reuse_b=*/false);
}

__device__ __forceinline__ unsigned short f2bf(float x) {
    unsigned u = __float_as_uint(x);
    u += 0x7FFFu + ((u >> 16) & 1u);   // round-to-nearest-even
    return (unsigned short)(u >> 16);
}

__device__ __forceinline__ float sigm(float x) { return 1.f / (1.f + __expf(-x)); }

// ---------------------------------------------------------------------------
// Workspace layout (bytes, all offsets 256B aligned; lifetime-aliased)
// ---------------------------------------------------------------------------
constexpr size_t O_IND   = 0;                       // 3 f32 (straight-through gates)
constexpr size_t O_HP    = 256;                     // 12x9 f32 head softmax
constexpr size_t O_HPS   = 768;                     // 12 f32 (sum_k hp)
constexpr size_t O_VB9   = 1024;                    // 64x9 f32 new_vb
constexpr size_t O_VB9S  = 3584;                    // 64 f32 sum_k new_vb
constexpr size_t O_STATS = 3840;                    // 2x64x2 f32 BN stats
constexpr size_t O_QKWB  = 5120;                    // qk_w bf16 (1536x768)
constexpr size_t O_VWB   = O_QKWB + 2359296;        // v_w bf16
constexpr size_t O_PWB   = O_VWB  + 1179648;        // proj_w bf16
constexpr size_t O_PW4B  = O_PWB  + 1179648;        // pw4 bf16 (768x64)
constexpr size_t O_PW9B  = O_PW4B + 98304;          // pw9 bf16
constexpr size_t O_C1PRE = O_PW9B + 98304;          // 12544x64 f32
constexpr size_t O_C3PRE = O_C1PRE + 3211264;       // 12544x64 f32
constexpr size_t O_C1BN  = O_C3PRE + 3211264;       // 12544x64 bf16
constexpr size_t O_C3BN  = O_C1BN + 1605632;        // 12544x64 bf16
constexpr size_t O_XB    = O_C3BN + 1605632;        // x bf16 -> later msa bf16
constexpr size_t O_VB16  = O_XB   + 19365888;       // (v_weight+v_b) bf16
constexpr size_t O_QKB   = O_VB16 + 19365888;       // qk bf16 -> later c1p f32
constexpr size_t O_VF    = O_QKB  + 38731776;       // v_weight f32 -> later c3p f32
constexpr size_t O_MSAP  = O_VF   + 38731776;       // msa proj f32
constexpr size_t O_NEWV  = O_MSAP + 38731776;       // 64x196x9x64 f32

// ---------------------------------------------------------------------------
// Small glue kernels
// ---------------------------------------------------------------------------
__global__ void k_scalars(const float* __restrict__ uni, const float* __restrict__ kth,
                          float* __restrict__ ind_ws, float* __restrict__ out_tail) {
    float ki_prod = 1.f;
    for (int i = 0; i < 3; ++i) {
        float kt = sigm(kth[i]);
        float p  = fminf(fmaxf(kt, EPSF), 1.f - EPSF);
        float u  = fminf(fmaxf(uni[i], EPSF), 1.f - EPSF);
        float lg = (__logf(u) - log1pf(-u) + __logf(p) - log1pf(-p)); // TEMP=1
        float ys = sigm(lg);
        float hard = (lg > 0.f) ? 1.f : 0.f;
        ind_ws[i] = ys + (hard - ys);
        float sgk = sigm(kt - 0.5f);
        float kh  = (kt > 0.5f) ? 1.f : 0.f;
        ki_prod *= (sgk + (kh - sgk));
        out_tail[i]     = ki_prod;  // ki (cumprod)
        out_tail[3 + i] = kt;       // kt
    }
}

__global__ void k_hp(const float* __restrict__ hp_in, float* __restrict__ hp,
                     float* __restrict__ hps) {
    int t = threadIdx.x;
    if (t < 9) {
        float mx = -3e38f;
        for (int h = 0; h < NH; ++h) mx = fmaxf(mx, hp_in[h * 9 + t] * (1.f / ALPHA));
        float s = 0.f;
        for (int h = 0; h < NH; ++h) s += __expf(hp_in[h * 9 + t] * (1.f / ALPHA) - mx);
        for (int h = 0; h < NH; ++h)
            hp[h * 9 + t] = __expf(hp_in[h * 9 + t] * (1.f / ALPHA) - mx) / s;
    }
    __syncthreads();
    if (t < NH) {
        float s = 0.f;
        for (int k = 0; k < 9; ++k) s += hp[t * 9 + k];
        hps[t] = s;
    }
}

__global__ void k_cast(const float* __restrict__ in, unsigned short* __restrict__ out, int n) {
    int i = blockIdx.x * blockDim.x + threadIdx.x;
    if (i < n) out[i] = f2bf(in[i]);
}

__global__ void k_vb(const float* __restrict__ v_b, const float* __restrict__ hp,
                     float* __restrict__ vb9, float* __restrict__ vb9s) {
    int d = threadIdx.x;
    if (d >= HD) return;
    float tot = 0.f;
    for (int k = 0; k < 9; ++k) {
        float s = 0.f;
        for (int h = 0; h < NH; ++h) s += v_b[h * HD + d] * hp[h * 9 + k];
        vb9[d * 9 + k] = s;
        tot += s;
    }
    vb9s[d] = tot;
}

__global__ void k_pw(const float* __restrict__ proj_w, const float* __restrict__ hp,
                     const float* __restrict__ hps, unsigned short* __restrict__ pw4b,
                     unsigned short* __restrict__ pw9b) {
    int i = blockIdx.x * blockDim.x + threadIdx.x;
    if (i >= DIM * HD) return;
    int c = i >> 6, d = i & 63;
    float s4 = 0.f, s9 = 0.f;
    for (int h = 0; h < NH; ++h) {
        float w = proj_w[(size_t)c * DIM + h * HD + d];
        s4 += w * hp[h * 9 + 4];
        s9 += w * hps[h];
    }
    pw4b[i] = f2bf(s4);
    pw9b[i] = f2bf(s9);
}

__global__ void k_newv(const float* __restrict__ vf, const float* __restrict__ hp,
                       const float* __restrict__ vb9, float* __restrict__ newv,
                       float* __restrict__ c1pre) {
    int i = blockIdx.x * blockDim.x + threadIdx.x;
    if (i >= CROWS * HD) return;
    int d = i & 63, bp = i >> 6;
    int b = bp / NP, p = bp % NP;
    size_t vrow = ((size_t)b * NTOK + 1 + p) * DIM;
    float vh[NH];
    for (int h = 0; h < NH; ++h) vh[h] = vf[vrow + h * HD + d];
    for (int kk = 0; kk < 9; ++kk) {
        float s = 0.f;
        for (int h = 0; h < NH; ++h) s += vh[h] * hp[h * 9 + kk];
        newv[((size_t)bp * 9 + kk) * HD + d] = s;
        if (kk == 4) c1pre[(size_t)bp * HD + d] = s + vb9[d * 9 + 4];
    }
}

__global__ void k_c3pre(const float* __restrict__ newv, const float* __restrict__ vb9s,
                        float* __restrict__ c3pre) {
    int i = blockIdx.x * blockDim.x + threadIdx.x;
    if (i >= CROWS * HD) return;
    int d = i & 63, bp = i >> 6;
    int b = bp / NP, q = bp % NP;
    int y = q / HGRID, x = q % HGRID;
    float acc = vb9s[d];
    for (int p = 0; p < 9; ++p) {
        int yy = y + (p % 3) - 1, xx = x + (p / 3) - 1;
        if (yy >= 0 && yy < HGRID && xx >= 0 && xx < HGRID)
            acc += newv[(((size_t)(b * NP + yy * HGRID + xx)) * 9 + p) * HD + d];
    }
    c3pre[(size_t)bp * HD + d] = acc;
}

__global__ void k_stats(const float* __restrict__ c1pre, const float* __restrict__ c3pre,
                        float* __restrict__ stats) {
    int blk = blockIdx.x;            // 0..127 : branch*64 + d
    int branch = blk >> 6, d = blk & 63;
    const float* src = branch ? c3pre : c1pre;
    float s = 0.f, sq = 0.f;
    for (int i = threadIdx.x; i < CROWS; i += blockDim.x) {
        float v = src[(size_t)i * HD + d];
        s += v; sq += v * v;
    }
    __shared__ float sh[256], sh2[256];
    sh[threadIdx.x] = s; sh2[threadIdx.x] = sq;
    __syncthreads();
    for (int off = 128; off > 0; off >>= 1) {
        if (threadIdx.x < off) { sh[threadIdx.x] += sh[threadIdx.x + off];
                                 sh2[threadIdx.x] += sh2[threadIdx.x + off]; }
        __syncthreads();
    }
    if (threadIdx.x == 0) {
        float mean = sh[0] * (1.f / CROWS);
        float var  = sh2[0] * (1.f / CROWS) - mean * mean;
        stats[blk * 2 + 0] = mean;
        stats[blk * 2 + 1] = rsqrtf(var + BN_EPS);
    }
}

__global__ void k_bnapply(const float* __restrict__ c1pre, const float* __restrict__ c3pre,
                          const float* __restrict__ stats,
                          const float* __restrict__ g1, const float* __restrict__ b1,
                          const float* __restrict__ g3, const float* __restrict__ b3,
                          unsigned short* __restrict__ c1bn, unsigned short* __restrict__ c3bn) {
    int i = blockIdx.x * blockDim.x + threadIdx.x;
    if (i >= CROWS * HD) return;
    int d = i & 63;
    float v1 = (c1pre[i] - stats[d * 2]) * stats[d * 2 + 1] * g1[d] + b1[d];
    float v3 = (c3pre[i] - stats[(64 + d) * 2]) * stats[(64 + d) * 2 + 1] * g3[d] + b3[d];
    c1bn[i] = f2bf(fmaxf(v1, 0.f));
    c3bn[i] = f2bf(fmaxf(v3, 0.f));
}

__global__ void k_combine(const float* __restrict__ x, const float* __restrict__ msap,
                          const float* __restrict__ c1p, const float* __restrict__ c3p,
                          const float* __restrict__ ind, float* __restrict__ out) {
    int i = blockIdx.x * blockDim.x + threadIdx.x;
    if (i >= OUTN) return;
    int c = i % DIM, row = i / DIM;
    int b = row / NTOK, n = row % NTOK;
    float i0 = ind[0], i1 = ind[1], i2 = ind[2];
    float m = msap[i], a, b3;
    if (n == 0) { a = b3 = x[i]; }
    else {
        size_t r = ((size_t)(b * NP + n - 1)) * DIM + c;
        a = c1p[r]; b3 = c3p[r];
    }
    out[i] = i2 * m + i1 * (1.f - i2) * b3 + i0 * (1.f - i1) * (1.f - i2) * a;
}

// ---------------------------------------------------------------------------
// GEMM: D[M,N] = A_bf16[M,K] * W_bf16[N,K]^T (+bias).  M%64==N%64==K%32==0.
// 128 threads = 4 waves, each wave computes a 32x32 tile (2x2 WMMA tiles).
// ---------------------------------------------------------------------------
__global__ void __launch_bounds__(128)
k_gemm(const unsigned short* __restrict__ A, const unsigned short* __restrict__ W,
       const float* __restrict__ biasF, const float* __restrict__ biasB,
       float* __restrict__ outF, unsigned short* __restrict__ outB,
       int M, int N, int K) {
    int wave = threadIdx.x >> 5, lane = threadIdx.x & 31;
    int half = lane >> 4, l16 = lane & 15;
    int m0 = blockIdx.x * 64 + (wave >> 1) * 32;
    int n0 = blockIdx.y * 64 + (wave & 1) * 32;

    v8f acc[2][2] = {};
    const unsigned short* a0p = A + (size_t)(m0 + l16) * K + (half ? 8 : 0);
    const unsigned short* a1p = A + (size_t)(m0 + 16 + l16) * K + (half ? 8 : 0);
    const unsigned short* b0p = W + (size_t)(n0 + l16) * K + (half ? 8 : 0);
    const unsigned short* b1p = W + (size_t)(n0 + 16 + l16) * K + (half ? 8 : 0);

    for (int kb = 0; kb < K; kb += 32) {
        Frag a0, a1, b0, b1;
        a0.q[0] = *(const v4u*)(a0p + kb);      a0.q[1] = *(const v4u*)(a0p + kb + 16);
        a1.q[0] = *(const v4u*)(a1p + kb);      a1.q[1] = *(const v4u*)(a1p + kb + 16);
        b0.q[0] = *(const v4u*)(b0p + kb);      b0.q[1] = *(const v4u*)(b0p + kb + 16);
        b1.q[0] = *(const v4u*)(b1p + kb);      b1.q[1] = *(const v4u*)(b1p + kb + 16);
        __builtin_prefetch(a0p + kb + 64, 0, 1);
        __builtin_prefetch(b0p + kb + 64, 0, 1);
        acc[0][0] = wmma_bf16(a0, b0, acc[0][0]);
        acc[0][1] = wmma_bf16(a0, b1, acc[0][1]);
        acc[1][0] = wmma_bf16(a1, b0, acc[1][0]);
        acc[1][1] = wmma_bf16(a1, b1, acc[1][1]);
    }

    for (int i = 0; i < 2; ++i)
        for (int j = 0; j < 2; ++j) {
            int col = n0 + j * 16 + l16;
            float bvF = biasF ? biasF[col] : 0.f;
            float bvB = biasB ? biasB[col] : 0.f;
#pragma unroll
            for (int e = 0; e < 8; ++e) {
                int row = m0 + i * 16 + half * 8 + e;
                size_t idx = (size_t)row * N + col;
                float v = acc[i][j][e];
                if (outF) outF[idx] = v + bvF;
                if (outB) outB[idx] = f2bf(v + bvB);
            }
        }
}

// ---------------------------------------------------------------------------
// Flash attention: one block per (b,h). 128 threads = 4 waves.
// Q/K fragments straight from global (bf16 row-major), V^T staged in LDS.
// Online softmax; P staged through LDS to form A-fragments (K=32 pairs).
// ---------------------------------------------------------------------------
__global__ void __launch_bounds__(128)
k_attn(const unsigned short* __restrict__ qk, const unsigned short* __restrict__ vmat,
       unsigned short* __restrict__ msa) {
    __shared__ alignas(16) unsigned short Vt[HD * 224];   // V^T, cols padded 197->224 w/ zeros
    __shared__ alignas(16) unsigned short Pst[4][16 * 32];

    int b = blockIdx.x / NH, h = blockIdx.x % NH;
    const unsigned short* Qb = qk + (size_t)b * NTOK * (2 * DIM) + h * HD;
    const unsigned short* Kb = Qb + DIM;
    const unsigned short* Vb = vmat + (size_t)b * NTOK * DIM + h * HD;
    unsigned short* Ob = msa + (size_t)b * NTOK * DIM + h * HD;

    int tid = threadIdx.x;
    for (int i = tid; i < 224 * HD; i += 128) {
        int n = i >> 6, d = i & 63;
        Vt[d * 224 + n] = (n < NTOK) ? Vb[(size_t)n * DIM + d] : (unsigned short)0;
    }
    __syncthreads();

    int wave = tid >> 5, lane = tid & 31, half = lane >> 4, l16 = lane & 15;

    for (int rt = wave; rt < 13; rt += 4) {            // 13 row tiles of 16
        int m0 = rt * 16;
        int qrow = m0 + l16; if (qrow > NTOK - 1) qrow = NTOK - 1;
        Frag aq0, aq1;
        const unsigned short* qp = Qb + (size_t)qrow * (2 * DIM) + (half ? 8 : 0);
        aq0.q[0] = *(const v4u*)(qp);        aq0.q[1] = *(const v4u*)(qp + 16);
        aq1.q[0] = *(const v4u*)(qp + 32);   aq1.q[1] = *(const v4u*)(qp + 48);

        float mrow[8], lrow[8];
        v8f oacc[4] = {};
#pragma unroll
        for (int e = 0; e < 8; ++e) { mrow[e] = -3e38f; lrow[e] = 0.f; }

        for (int ct2 = 0; ct2 < 7; ++ct2) {            // column tiles in pairs (K=32)
            int n0 = ct2 * 32;
            v8f s0 = {}, s1 = {};
            {
                int nc0 = n0 + l16;      if (nc0 > NTOK - 1) nc0 = NTOK - 1;
                int nc1 = n0 + 16 + l16; if (nc1 > NTOK - 1) nc1 = NTOK - 1;
                const unsigned short* kp0 = Kb + (size_t)nc0 * (2 * DIM) + (half ? 8 : 0);
                const unsigned short* kp1 = Kb + (size_t)nc1 * (2 * DIM) + (half ? 8 : 0);
                Frag bk;
                bk.q[0] = *(const v4u*)(kp0);      bk.q[1] = *(const v4u*)(kp0 + 16);
                s0 = wmma_bf16(aq0, bk, s0);
                bk.q[0] = *(const v4u*)(kp0 + 32); bk.q[1] = *(const v4u*)(kp0 + 48);
                s0 = wmma_bf16(aq1, bk, s0);
                bk.q[0] = *(const v4u*)(kp1);      bk.q[1] = *(const v4u*)(kp1 + 16);
                s1 = wmma_bf16(aq0, bk, s1);
                bk.q[0] = *(const v4u*)(kp1 + 32); bk.q[1] = *(const v4u*)(kp1 + 48);
                s1 = wmma_bf16(aq1, bk, s1);
            }
            bool v0 = (n0 + l16) < NTOK;
            bool v1 = (n0 + 16 + l16) < NTOK;
#pragma unroll
            for (int e = 0; e < 8; ++e) {
                float a0 = v0 ? s0[e] * SCALE : -3e38f;
                float a1 = v1 ? s1[e] * SCALE : -3e38f;
                float mx = fmaxf(a0, a1);
                mx = fmaxf(mx, __shfl_xor(mx, 1));
                mx = fmaxf(mx, __shfl_xor(mx, 2));
                mx = fmaxf(mx, __shfl_xor(mx, 4));
                mx = fmaxf(mx, __shfl_xor(mx, 8));
                float mnew = fmaxf(mrow[e], mx);
                float p0 = __expf(a0 - mnew);
                float p1 = __expf(a1 - mnew);
                float rs = p0 + p1;
                rs += __shfl_xor(rs, 1);
                rs += __shfl_xor(rs, 2);
                rs += __shfl_xor(rs, 4);
                rs += __shfl_xor(rs, 8);
                float alpha = __expf(mrow[e] - mnew);
                lrow[e] = lrow[e] * alpha + rs;
                mrow[e] = mnew;
                oacc[0][e] = oacc[0][e] * alpha;
                oacc[1][e] = oacc[1][e] * alpha;
                oacc[2][e] = oacc[2][e] * alpha;
                oacc[3][e] = oacc[3][e] * alpha;
                int r = half * 8 + e;                  // C-layout row within tile
                Pst[wave][r * 32 + l16]      = f2bf(p0);
                Pst[wave][r * 32 + 16 + l16] = f2bf(p1);
            }
            // LDS ops are in-order per wave; compiler inserts s_wait_dscnt.
            Frag aP;
            {
                const unsigned short* pp = &Pst[wave][l16 * 32 + (half ? 8 : 0)];
                aP.q[0] = *(const v4u*)(pp);
                aP.q[1] = *(const v4u*)(pp + 16);
            }
#pragma unroll
            for (int t = 0; t < 4; ++t) {
                const unsigned short* vp = &Vt[(t * 16 + l16) * 224 + n0 + (half ? 8 : 0)];
                Frag bV;
                bV.q[0] = *(const v4u*)(vp);
                bV.q[1] = *(const v4u*)(vp + 16);
                oacc[t] = wmma_bf16(aP, bV, oacc[t]);
            }
        }
#pragma unroll
        for (int t = 0; t < 4; ++t)
#pragma unroll
            for (int e = 0; e < 8; ++e) {
                int r = m0 + half * 8 + e;
                if (r < NTOK) {
                    float val = oacc[t][e] / lrow[e];
                    Ob[(size_t)r * DIM + t * 16 + l16] = f2bf(val);
                }
            }
    }
}

// ---------------------------------------------------------------------------
// Launcher
// ---------------------------------------------------------------------------
extern "C" void kernel_launch(void* const* d_in, const int* in_sizes, int n_in,
                              void* d_out, int out_size, void* d_ws, size_t ws_size,
                              hipStream_t stream) {
    (void)in_sizes; (void)n_in; (void)out_size; (void)ws_size;
    const float* x    = (const float*)d_in[0];
    const float* uni  = (const float*)d_in[1];
    const float* qkw  = (const float*)d_in[2];
    const float* qkb  = (const float*)d_in[3];
    const float* vw   = (const float*)d_in[4];
    const float* vb   = (const float*)d_in[5];
    const float* pw   = (const float*)d_in[6];
    const float* pb   = (const float*)d_in[7];
    const float* g1   = (const float*)d_in[8];
    const float* b1   = (const float*)d_in[9];
    const float* g3   = (const float*)d_in[10];
    const float* b3   = (const float*)d_in[11];
    const float* kth  = (const float*)d_in[12];
    const float* hpin = (const float*)d_in[13];
    float* out = (float*)d_out;
    char* w = (char*)d_ws;

    float* indw   = (float*)(w + O_IND);
    float* hpw    = (float*)(w + O_HP);
    float* hpsw   = (float*)(w + O_HPS);
    float* vb9w   = (float*)(w + O_VB9);
    float* vb9sw  = (float*)(w + O_VB9S);
    float* statsw = (float*)(w + O_STATS);
    unsigned short* qkwb  = (unsigned short*)(w + O_QKWB);
    unsigned short* vwb   = (unsigned short*)(w + O_VWB);
    unsigned short* pwb   = (unsigned short*)(w + O_PWB);
    unsigned short* pw4b  = (unsigned short*)(w + O_PW4B);
    unsigned short* pw9b  = (unsigned short*)(w + O_PW9B);
    float*          c1pre = (float*)(w + O_C1PRE);
    float*          c3pre = (float*)(w + O_C3PRE);
    unsigned short* c1bn  = (unsigned short*)(w + O_C1BN);
    unsigned short* c3bn  = (unsigned short*)(w + O_C3BN);
    unsigned short* xb    = (unsigned short*)(w + O_XB);    // later: msa bf16
    unsigned short* msab  = (unsigned short*)(w + O_XB);
    unsigned short* vb16  = (unsigned short*)(w + O_VB16);
    unsigned short* qkb16 = (unsigned short*)(w + O_QKB);   // later: c1p f32
    float*          c1p   = (float*)(w + O_QKB);
    float*          vf    = (float*)(w + O_VF);             // later: c3p f32
    float*          c3p   = (float*)(w + O_VF);
    float*          msap  = (float*)(w + O_MSAP);
    float*          newv  = (float*)(w + O_NEWV);

    k_scalars<<<1, 1, 0, stream>>>(uni, kth, indw, out + OUTN);
    k_hp<<<1, 32, 0, stream>>>(hpin, hpw, hpsw);

    k_cast<<<(OUTN + 255) / 256, 256, 0, stream>>>(x, xb, OUTN);
    k_cast<<<(2 * DIM * DIM + 255) / 256, 256, 0, stream>>>(qkw, qkwb, 2 * DIM * DIM);
    k_cast<<<(DIM * DIM + 255) / 256, 256, 0, stream>>>(vw, vwb, DIM * DIM);
    k_cast<<<(DIM * DIM + 255) / 256, 256, 0, stream>>>(pw, pwb, DIM * DIM);

    dim3 blk(128);
    // qk = x @ qk_w^T + qk_b  -> bf16 (12608 x 1536)
    k_gemm<<<dim3(ROWS / 64, (2 * DIM) / 64), blk, 0, stream>>>(
        xb, qkwb, nullptr, qkb, nullptr, qkb16, ROWS, 2 * DIM, DIM);
    // v_weight = x @ v_w^T -> f32 (conv path);  v_weight + v_b -> bf16 (attn path)
    k_gemm<<<dim3(ROWS / 64, DIM / 64), blk, 0, stream>>>(
        xb, vwb, nullptr, vb, vf, vb16, ROWS, DIM, DIM);
    // attention -> msa bf16 (reuses xb region)
    k_attn<<<BSZ * NH, 128, 0, stream>>>(qkb16, vb16, msab);
    // msa @ proj_w^T + proj_b -> f32
    k_gemm<<<dim3(ROWS / 64, DIM / 64), blk, 0, stream>>>(
        msab, pwb, pb, nullptr, msap, nullptr, ROWS, DIM, DIM);

    // conv branches
    k_vb<<<1, 64, 0, stream>>>(vb, hpw, vb9w, vb9sw);
    k_pw<<<(DIM * HD + 255) / 256, 256, 0, stream>>>(pw, hpw, hpsw, pw4b, pw9b);
    k_newv<<<(CROWS * HD + 255) / 256, 256, 0, stream>>>(vf, hpw, vb9w, newv, c1pre);
    k_c3pre<<<(CROWS * HD + 255) / 256, 256, 0, stream>>>(newv, vb9sw, c3pre);
    k_stats<<<128, 256, 0, stream>>>(c1pre, c3pre, statsw);
    k_bnapply<<<(CROWS * HD + 255) / 256, 256, 0, stream>>>(
        c1pre, c3pre, statsw, g1, b1, g3, b3, c1bn, c3bn);
    k_gemm<<<dim3(CROWS / 64, DIM / 64), blk, 0, stream>>>(
        c1bn, pw4b, pb, nullptr, c1p, nullptr, CROWS, DIM, HD);
    k_gemm<<<dim3(CROWS / 64, DIM / 64), blk, 0, stream>>>(
        c3bn, pw9b, pb, nullptr, c3p, nullptr, CROWS, DIM, HD);

    k_combine<<<(OUTN + 255) / 256, 256, 0, stream>>>(x, msap, c1p, c3p, indw, out);
}
